// SubnetGate_58634893525694
// MI455X (gfx1250) — matrix operand order repaired and checked by reading, not compiled, not run on hardware
//
#include <hip/hip_runtime.h>

// ---------------- problem constants ----------------
constexpr int B_    = 8192;
constexpr int DIN   = 512;
constexpr int DH    = 2048;
constexpr int DOUT  = 512;
constexpr int NEXP  = 8;

constexpr int MTILE = 32;                 // rows per GEMM block
constexpr int NBLK  = B_ / MTILE + NEXP;  // 264 max row-blocks (expert padding)
constexpr int RT16  = NBLK * 2;           // 528 16-row tiles
constexpr int KT1   = DIN / 32;           // 16  k-tiles, stage 1
constexpr int NT1   = DH / 16;            // 128 n-tiles, stage 1
constexpr int KT2   = DH / 32;            // 64  k-tiles, stage 2
constexpr int NT2   = DOUT / 16;          // 32  n-tiles, stage 2
constexpr int PERM_N = NBLK * MTILE;      // 8448

// ---------------- workspace layout (bytes) ----------------
constexpr size_t OFF_COUNTS  = 0;
constexpr size_t OFF_CURSORS = 64;
constexpr size_t OFF_TILEEXP = 128;                                    // NBLK ints
constexpr size_t OFF_PERM    = 1280;                                   // PERM_N uints
constexpr size_t OFF_XG      = 35072;                                  // RT16*KT1*512 bf16
constexpr size_t XG_BYTES    = (size_t)RT16 * KT1 * 512 * 2;
constexpr size_t OFF_W1F     = OFF_XG + XG_BYTES;                      // 8,685,824
constexpr size_t W1F_BYTES   = (size_t)NEXP * NT1 * KT1 * 512 * 2;     // 16 MB
constexpr size_t OFF_W2F     = OFF_W1F + W1F_BYTES;
constexpr size_t W2F_BYTES   = (size_t)NEXP * NT2 * KT2 * 512 * 2;     // 16 MB
constexpr size_t WS_NEEDED   = OFF_W2F + W2F_BYTES;                    // ~42.2 MB

typedef __attribute__((ext_vector_type(16))) __bf16 v16bf;
typedef __attribute__((ext_vector_type(8)))  float  v8f;

__device__ __forceinline__ unsigned short f2bf(float f) {
    unsigned u = __builtin_bit_cast(unsigned, f);
    unsigned r = u + 0x7fffu + ((u >> 16) & 1u);   // round to nearest even
    return (unsigned short)(r >> 16);
}

// ---------------- routing kernels ----------------
__global__ void k_init(int* counts) {
    if (threadIdx.x < NEXP) counts[threadIdx.x] = 0;
}

__global__ void k_fill_perm(unsigned* perm) {
    int i = blockIdx.x * 256 + threadIdx.x;
    if (i < PERM_N) perm[i] = 0xFFFFFFFFu;
}

__global__ void k_hist(const long long* __restrict__ groups, int* counts) {
    int i = blockIdx.x * 256 + threadIdx.x;
    if (i < B_) atomicAdd(&counts[(int)groups[i]], 1);
}

__global__ void k_prefix(const int* __restrict__ counts, int* cursors, int* tileExpert) {
    // single thread: tiny E=8 scan + tile table
    int run = 0, tb = 0;
    for (int e = 0; e < NEXP; ++e) {
        cursors[e] = run;
        int nt = (counts[e] + MTILE - 1) / MTILE;
        for (int k = 0; k < nt; ++k) tileExpert[tb++] = e;
        run += nt * MTILE;
    }
    for (; tb < NBLK; ++tb) tileExpert[tb] = -1;
}

__global__ void k_scatter(const long long* __restrict__ groups, int* cursors,
                          unsigned* __restrict__ perm) {
    int i = blockIdx.x * 256 + threadIdx.x;
    if (i < B_) {
        int e = (int)groups[i];
        int slot = atomicAdd(&cursors[e], 1);
        perm[slot] = (unsigned)i;
    }
}

// ---------------- x gather + bf16 + A-fragment pack ----------------
// A-frag (16-bit 16x32): lane = m + 16*bit3(k'), v = 4*bit4(k') + bits2:1(k'), h = bit0(k')
__global__ void k_gather_x(const float* __restrict__ x, const unsigned* __restrict__ perm,
                           unsigned short* __restrict__ xg) {
    __shared__ unsigned short sh[512];
    int rt16 = blockIdx.x >> 4;
    int kt   = blockIdx.x & 15;
    int t = threadIdx.x;
#pragma unroll
    for (int p = 0; p < 2; ++p) {
        int idx = t + p * 256;
        int m  = idx >> 5;
        int kp = idx & 31;
        unsigned row = perm[rt16 * 16 + m];
        float v = 0.0f;
        if (row != 0xFFFFFFFFu) v = x[(size_t)row * DIN + kt * 32 + kp];
        int lane = m + 16 * ((kp >> 3) & 1);
        int vr   = 4 * ((kp >> 4) & 1) + ((kp >> 1) & 3);
        int hh   = kp & 1;
        sh[lane * 16 + vr * 2 + hh] = f2bf(v);
    }
    __syncthreads();
    ((unsigned*)xg)[(size_t)blockIdx.x * 256 + t] = ((const unsigned*)sh)[t];
}

// ---------------- weight bf16 + B-fragment pack ----------------
// B-frag (16-bit 32x16): lane = n + 16*bit3(k), v = 4*bit4(k) + bits2:1(k), h = bit0(k)
__global__ void k_conv_w(const float* __restrict__ W, unsigned short* __restrict__ Wf,
                         int NT, int KT, int srcK, int srcN) {
    __shared__ unsigned short sh[512];
    int f  = blockIdx.x;
    int e  = f / (NT * KT);
    int r  = f % (NT * KT);
    int nt = r / KT;
    int kt = r % KT;
    int t = threadIdx.x;
#pragma unroll
    for (int p = 0; p < 2; ++p) {
        int idx = t + p * 256;
        int k = idx >> 4;
        int n = idx & 15;
        float v = W[((size_t)e * srcK + (size_t)kt * 32 + k) * srcN + (size_t)nt * 16 + n];
        int lane = n + 16 * ((k >> 3) & 1);
        int vr   = 4 * ((k >> 4) & 1) + ((k >> 1) & 3);
        int hh   = k & 1;
        sh[lane * 16 + vr * 2 + hh] = f2bf(v);
    }
    __syncthreads();
    ((unsigned*)Wf)[(size_t)f * 256 + t] = ((const unsigned*)sh)[t];
}

// ---------------- fused MoE MLP GEMM (bf16 WMMA, f32 accum) ----------------
__global__ void __launch_bounds__(256) k_gemm(
        const unsigned short* __restrict__ xg,
        const unsigned short* __restrict__ W1f,
        const unsigned short* __restrict__ W2f,
        const float* __restrict__ b1, const float* __restrict__ b2,
        const unsigned* __restrict__ perm, const int* __restrict__ tileExpert,
        float* __restrict__ out) {
    int blk = blockIdx.x;
    int e = tileExpert[blk];
    if (e < 0) return;

    int w    = threadIdx.x >> 5;
    int lane = threadIdx.x & 31;

    __shared__ unsigned short shx[2 * KT1 * 512];  // 32 KB: x A-frags, 2 row-tiles
    __shared__ unsigned short shh[2 * 8 * 512];    // 16 KB: h A-frags per 256-chunk

    // stage x fragments into LDS once
    {
        const unsigned* src = (const unsigned*)xg + (size_t)blk * 2 * KT1 * 256;
        unsigned* dst = (unsigned*)shx;
        for (int i = threadIdx.x; i < 2 * KT1 * 256; i += 256) dst[i] = src[i];
    }
    __syncthreads();

    v8f acc[2][4];
#pragma unroll
    for (int rt = 0; rt < 2; ++rt)
#pragma unroll
        for (int j = 0; j < 4; ++j) acc[rt][j] = (v8f){0,0,0,0,0,0,0,0};

    int n     = lane & 15;
    int mh    = (lane >> 4) << 3;
    int dlane = ((n >> 3) & 1) << 4;
    int nsub  = (n >> 1) & 3;
    int nbit0 = n & 1;

    for (int c = 0; c < 8; ++c) {          // D_H chunks of 256
        // ---- stage 1: h[32 x 256] = relu(x * W1 + b1), packed to LDS A-frags ----
        v8f hacc[2][2];
#pragma unroll
        for (int j = 0; j < 2; ++j)
#pragma unroll
            for (int rt = 0; rt < 2; ++rt) hacc[j][rt] = (v8f){0,0,0,0,0,0,0,0};

#pragma unroll
        for (int kt = 0; kt < KT1; ++kt) {
            v16bf a0 = *(const v16bf*)(shx + ((0 * KT1 + kt) << 9) + lane * 16);
            v16bf a1 = *(const v16bf*)(shx + ((1 * KT1 + kt) << 9) + lane * 16);
#pragma unroll
            for (int j = 0; j < 2; ++j) {
                int ntH = c * 16 + w * 2 + j;
                const unsigned short* bp =
                    W1f + ((((size_t)e * NT1 + ntH) * KT1 + kt) << 9) + lane * 16;
                if (kt + 1 < KT1) __builtin_prefetch(bp + 512, 0, 1);
                v16bf bf = *(const v16bf*)bp;
                hacc[j][0] = __builtin_amdgcn_wmma_f32_16x16x32_bf16(
                    false, a0, false, bf, (short)0, hacc[j][0], false, false);
                hacc[j][1] = __builtin_amdgcn_wmma_f32_16x16x32_bf16(
                    false, a1, false, bf, (short)0, hacc[j][1], false, false);
            }
        }
#pragma unroll
        for (int j = 0; j < 2; ++j) {
            int ntH = c * 16 + w * 2 + j;
            float bias = b1[(size_t)e * DH + ntH * 16 + n];
            int halfoff = ((j << 2) + nsub) * 2 + nbit0;   // v*2 + h for k' = j*16+n
#pragma unroll
            for (int rt = 0; rt < 2; ++rt) {
                unsigned short* dst = shh + ((rt * 8 + w) << 9);
#pragma unroll
                for (int r = 0; r < 8; ++r) {
                    float v = fmaxf(hacc[j][rt][r] + bias, 0.0f);
                    dst[(r + mh + dlane) * 16 + halfoff] = f2bf(v);
                }
            }
        }
        __syncthreads();

        // ---- stage 2: acc += h * W2 ----
#pragma unroll
        for (int ks = 0; ks < 8; ++ks) {
            v16bf ha0 = *(const v16bf*)(shh + ((0 * 8 + ks) << 9) + lane * 16);
            v16bf ha1 = *(const v16bf*)(shh + ((1 * 8 + ks) << 9) + lane * 16);
            int kt2 = c * 8 + ks;
#pragma unroll
            for (int j = 0; j < 4; ++j) {
                int ntO = w * 4 + j;
                v16bf bf = *(const v16bf*)(
                    W2f + ((((size_t)e * NT2 + ntO) * KT2 + kt2) << 9) + lane * 16);
                acc[0][j] = __builtin_amdgcn_wmma_f32_16x16x32_bf16(
                    false, ha0, false, bf, (short)0, acc[0][j], false, false);
                acc[1][j] = __builtin_amdgcn_wmma_f32_16x16x32_bf16(
                    false, ha1, false, bf, (short)0, acc[1][j], false, false);
            }
        }
        __syncthreads();
    }

    // ---- epilogue: + b2, scatter rows back ----
    unsigned rows[2][8];
#pragma unroll
    for (int rt = 0; rt < 2; ++rt)
#pragma unroll
        for (int r = 0; r < 8; ++r)
            rows[rt][r] = perm[blk * MTILE + rt * 16 + r + mh];

#pragma unroll
    for (int j = 0; j < 4; ++j) {
        int col = (w * 4 + j) * 16 + n;
        float bias2 = b2[(size_t)e * DOUT + col];
#pragma unroll
        for (int rt = 0; rt < 2; ++rt)
#pragma unroll
            for (int r = 0; r < 8; ++r) {
                unsigned row = rows[rt][r];
                if (row != 0xFFFFFFFFu)
                    out[(size_t)row * DOUT + col] = acc[rt][j][r] + bias2;
            }
    }
}

// ---------------- naive fallback (workspace too small) ----------------
__global__ void k_naive(const float* __restrict__ x, const long long* __restrict__ g,
                        const float* __restrict__ W1, const float* __restrict__ b1,
                        const float* __restrict__ W2, const float* __restrict__ b2,
                        float* __restrict__ out) {
    int row = blockIdx.x;
    int e = (int)g[row];
    __shared__ float h[DH];
    const float* xr = x + (size_t)row * DIN;
    for (int i = threadIdx.x; i < DH; i += 256) {
        float s = b1[(size_t)e * DH + i];
        const float* wp = W1 + (size_t)e * DIN * DH + i;
        for (int k = 0; k < DIN; ++k) s += xr[k] * wp[(size_t)k * DH];
        h[i] = fmaxf(s, 0.0f);
    }
    __syncthreads();
    for (int i = threadIdx.x; i < DOUT; i += 256) {
        float s = b2[(size_t)e * DOUT + i];
        const float* wp = W2 + (size_t)e * DH * DOUT + i;
        for (int k = 0; k < DH; ++k) s += h[k] * wp[(size_t)k * DOUT];
        out[(size_t)row * DOUT + i] = s;
    }
}

extern "C" void kernel_launch(void* const* d_in, const int* in_sizes, int n_in,
                              void* d_out, int out_size, void* d_ws, size_t ws_size,
                              hipStream_t stream) {
    const float*     x      = (const float*)d_in[0];
    const long long* groups = (const long long*)d_in[1];
    const float*     W1     = (const float*)d_in[2];
    const float*     b1     = (const float*)d_in[3];
    const float*     W2     = (const float*)d_in[4];
    const float*     b2     = (const float*)d_in[5];
    float*           out    = (float*)d_out;

    if (ws_size < WS_NEEDED) {
        k_naive<<<B_, 256, 0, stream>>>(x, groups, W1, b1, W2, b2, out);
        return;
    }

    char* w = (char*)d_ws;
    int*            counts     = (int*)(w + OFF_COUNTS);
    int*            cursors    = (int*)(w + OFF_CURSORS);
    int*            tileExpert = (int*)(w + OFF_TILEEXP);
    unsigned*       perm       = (unsigned*)(w + OFF_PERM);
    unsigned short* xg         = (unsigned short*)(w + OFF_XG);
    unsigned short* w1f        = (unsigned short*)(w + OFF_W1F);
    unsigned short* w2f        = (unsigned short*)(w + OFF_W2F);

    k_init<<<1, 32, 0, stream>>>(counts);
    k_fill_perm<<<(PERM_N + 255) / 256, 256, 0, stream>>>(perm);
    k_hist<<<B_ / 256, 256, 0, stream>>>(groups, counts);
    k_prefix<<<1, 1, 0, stream>>>(counts, cursors, tileExpert);
    k_scatter<<<B_ / 256, 256, 0, stream>>>(groups, cursors, perm);
    k_gather_x<<<RT16 * KT1, 256, 0, stream>>>(x, perm, xg);
    k_conv_w<<<NEXP * NT1 * KT1, 256, 0, stream>>>(W1, w1f, NT1, KT1, DIN, DH);
    k_conv_w<<<NEXP * NT2 * KT2, 256, 0, stream>>>(W2, w2f, NT2, KT2, DH, DOUT);
    k_gemm<<<NBLK, 256, 0, stream>>>(xg, w1f, w2f, b1, b2, perm, tileExpert, out);
}
